// MultiHeadAttention_78331613544927
// MI455X (gfx1250) — matrix operand order repaired and checked by reading, not compiled, not run on hardware
//
#include <hip/hip_runtime.h>
#include <hip/hip_bf16.h>
#include <math.h>
#include <stdint.h>

typedef __attribute__((ext_vector_type(16))) _Float16 v16h;
typedef __attribute__((ext_vector_type(8)))  _Float16 v8h;
typedef __attribute__((ext_vector_type(4)))  _Float16 v4h;
typedef __attribute__((ext_vector_type(8)))  float    v8f;

#define DMODEL 1024
#define NHEAD  16
#define HD     64
#define BATCH  4
#define SEQ    2048
#define MTOT   (BATCH*SEQ)   // 8192

// ---------------------------------------------------------------------------
// WMMA fragment helpers (CDNA5 16x16x32 f16, wave32)
// A-frag (16x32, MxK): lane L: row = L&15, hi = L>>4;
//   elems 0..7  -> K = hi*8 + e         (VGPR 0-3)
//   elems 8..15 -> K = 16 + hi*8 + e-8  (VGPR 4-7)
// B-frag (32x16, KxN): lane L: col n = L&15, hi = L>>4; same K order as A-frag
// C/D (16x16 f32): lane L: col = L&15, elem v -> row = v + (L>>4)*8
// ---------------------------------------------------------------------------
__device__ __forceinline__ v16h frag_a_ld(const _Float16* __restrict__ base, int ld, int lane) {
  const int row = lane & 15, hi = lane >> 4;
  const _Float16* p = base + row * ld + hi * 8;
  v8h lo = *(const v8h*)(p);
  v8h hv = *(const v8h*)(p + 16);
  v16h r;
#pragma unroll
  for (int i = 0; i < 8; ++i) { r[i] = lo[i]; r[i + 8] = hv[i]; }
  return r;
}

__device__ __forceinline__ v16h frag_b_ld(const _Float16* __restrict__ base, int ld, int lane) {
  const int n = lane & 15, hi = lane >> 4;
  const _Float16* p = base + n * ld + hi * 8;
  v8h lo = *(const v8h*)(p);
  v8h hv = *(const v8h*)(p + 16);
  v16h r;
#pragma unroll
  for (int i = 0; i < 8; ++i) { r[i] = lo[i]; r[i + 8] = hv[i]; }
  return r;
}

__device__ __forceinline__ v8f wmma16(v16h a, v16h b, v8f c) {
  return __builtin_amdgcn_wmma_f32_16x16x32_f16(false, a, false, b, (short)0, c, false, false);
}

// CDNA5 async copy: global -> LDS, 16 bytes per lane, tracked by ASYNCcnt.
// VDST carries the per-lane LDS byte address; VADDR the 64-bit global address.
__device__ __forceinline__ void async_ld_b128(const _Float16* lds_dst, const _Float16* gsrc) {
  unsigned lds_off = (unsigned)(uintptr_t)lds_dst;   // low 32 bits = LDS offset
  unsigned long long ga = (unsigned long long)(uintptr_t)gsrc;
  asm volatile("global_load_async_to_lds_b128 %0, %1, off"
               :: "v"(lds_off), "v"(ga) : "memory");
}
__device__ __forceinline__ void wait_async0() {
  asm volatile("s_wait_asynccnt 0x0" ::: "memory");
}
__device__ __forceinline__ void wait_ds0() {
  asm volatile("s_wait_dscnt 0x0" ::: "memory");
}

// ---------------------------------------------------------------------------
// Kernel 1: projection GEMM  out[m][n] = sum_k X[m][k] * W[n][k]
// M=8192 (b*s), N=1024 (h*d), K=1024. f32 in, f16 out in head-major layouts.
// mode 0: out[b,h,s,d] (Q with scale, K with scale=1)
// mode 2: out[b,h,d,s] (V transposed for attention B-frags)
// ---------------------------------------------------------------------------
#define G_BM 128
#define G_BN 128
#define G_BK 64
#define G_LD 72    // 64 + 8 halves padding

__global__ __launch_bounds__(256)
void proj_gemm_kernel(const float* __restrict__ X, const float* __restrict__ W,
                      _Float16* __restrict__ out, int mode, float scale)
{
  __shared__ alignas(16) _Float16 sA[G_BM * G_LD];
  __shared__ alignas(16) _Float16 sB[G_BN * G_LD];

  const int tid  = threadIdx.x;
  const int lane = tid & 31;
  const int w    = tid >> 5;       // 8 waves
  const int wm   = w & 3;          // 4 wave-rows (32 M each)
  const int wn   = w >> 2;         // 2 wave-cols (64 N each)
  const int m0   = blockIdx.x * G_BM;
  const int n0   = blockIdx.y * G_BN;

  v8f acc[2][4];
#pragma unroll
  for (int i = 0; i < 2; ++i)
#pragma unroll
    for (int j = 0; j < 4; ++j)
#pragma unroll
      for (int v = 0; v < 8; ++v) acc[i][j][v] = 0.0f;

  for (int k0 = 0; k0 < DMODEL; k0 += G_BK) {
    // stage 128x64 f32 -> f16 for A and W (each thread: 8 float4 per matrix)
#pragma unroll
    for (int i = 0; i < 8; ++i) {
      const int idx = tid + 256 * i;        // 0..2047 float4 chunks
      const int row = idx >> 4;             // 16 float4 per row
      const int c4  = idx & 15;
      const float4 a = *(const float4*)(X + (long)(m0 + row) * DMODEL + k0 + c4 * 4);
      const float4 b = *(const float4*)(W + (long)(n0 + row) * DMODEL + k0 + c4 * 4);
      v4h ah = { (_Float16)a.x, (_Float16)a.y, (_Float16)a.z, (_Float16)a.w };
      v4h bh = { (_Float16)b.x, (_Float16)b.y, (_Float16)b.z, (_Float16)b.w };
      *(v4h*)(&sA[row * G_LD + c4 * 4]) = ah;
      *(v4h*)(&sB[row * G_LD + c4 * 4]) = bh;
      if (k0 + G_BK < DMODEL) {
        __builtin_prefetch(X + (long)(m0 + row) * DMODEL + k0 + G_BK + c4 * 4, 0, 3);
        __builtin_prefetch(W + (long)(n0 + row) * DMODEL + k0 + G_BK + c4 * 4, 0, 3);
      }
    }
    __syncthreads();

#pragma unroll
    for (int ks = 0; ks < 2; ++ks) {
      const v16h a0 = frag_a_ld(sA + (wm * 32 +  0) * G_LD + ks * 32, G_LD, lane);
      const v16h a1 = frag_a_ld(sA + (wm * 32 + 16) * G_LD + ks * 32, G_LD, lane);
#pragma unroll
      for (int j = 0; j < 4; ++j) {
        const v16h bf = frag_b_ld(sB + (wn * 64 + j * 16) * G_LD + ks * 32, G_LD, lane);
        acc[0][j] = wmma16(a0, bf, acc[0][j]);
        acc[1][j] = wmma16(a1, bf, acc[1][j]);
      }
    }
    __syncthreads();
  }

  // epilogue: f32 -> f16, scatter to head-major layout
  const int hi = lane >> 4, nl = lane & 15;
#pragma unroll
  for (int i = 0; i < 2; ++i)
#pragma unroll
    for (int j = 0; j < 4; ++j)
#pragma unroll
      for (int v = 0; v < 8; ++v) {
        const int gm = m0 + wm * 32 + i * 16 + hi * 8 + v;
        const int gn = n0 + wn * 64 + j * 16 + nl;
        const float val = acc[i][j][v] * scale;
        const int b = gm >> 11;          // / SEQ
        const int s = gm & (SEQ - 1);
        const int h = gn >> 6;           // / HD
        const int d = gn & (HD - 1);
        long idx;
        if (mode == 2) idx = (((long)(b * NHEAD + h) * HD + d) * SEQ + s);
        else           idx = (((long)(b * NHEAD + h) * SEQ + s) * HD + d);
        out[idx] = (_Float16)val;
      }
}

// ---------------------------------------------------------------------------
// Kernel 2: flash attention per (b, h, 64-row q tile). 4 waves x 16 q-rows.
// Q,K: [b,h,s,d] f16 (Q pre-scaled by 1/sqrt(hd)); Vt: [b,h,d,s] f16.
// O: [b,s,h,d] f16 (= row-major (B,S,D) for the output projection).
// K/V chunks staged with CDNA5 async global->LDS copies (ASYNCcnt).
// ---------------------------------------------------------------------------
#define ALD 72      // 64 + 8 halves padding
#define SLD 68      // 64 + 4 f32 padding

__global__ __launch_bounds__(128)
void attn_kernel(const _Float16* __restrict__ Q, const _Float16* __restrict__ K,
                 const _Float16* __restrict__ Vt, const int* __restrict__ mask,
                 _Float16* __restrict__ O)
{
  __shared__ alignas(16) _Float16 sK[64 * ALD];
  __shared__ alignas(16) _Float16 sV[64 * ALD];
  __shared__ alignas(16) _Float16 sP[64 * ALD];
  __shared__ float sS[64 * SLD];
  __shared__ float sM[64], sL[64], sAl[64], sMask[64];

  const int tid  = threadIdx.x;
  const int lane = tid & 31;
  const int w    = tid >> 5;          // 4 waves
  const int q0   = blockIdx.x * 64;
  const int bh   = blockIdx.y;        // b*16 + h
  const int b    = bh >> 4;
  const int h    = bh & 15;
  const long hb  = (long)bh;

  const _Float16* qp = Q  + (hb * SEQ + q0) * HD;
  const _Float16* kp = K  + hb * SEQ * HD;
  const _Float16* vp = Vt + hb * HD * SEQ;
  const int*      mp = mask + b * SEQ;

  // Q fragments (16 rows x 64 d per wave; 2 k-steps) straight from global
  const v16h aq0 = frag_a_ld(qp + (w * 16) * HD,      HD, lane);
  const v16h aq1 = frag_a_ld(qp + (w * 16) * HD + 32, HD, lane);

  v8f o[4];
#pragma unroll
  for (int j = 0; j < 4; ++j)
#pragma unroll
    for (int v = 0; v < 8; ++v) o[j][v] = 0.0f;

  if (tid < 64) { sM[tid] = -3.0e38f; sL[tid] = 0.0f; }

  const int hi = lane >> 4, nl = lane & 15;

  for (int kc = 0; kc < SEQ / 64; ++kc) {
    const int kk = kc * 64;
    __syncthreads();   // previous chunk's compute done before restaging
#pragma unroll
    for (int i = 0; i < 4; ++i) {
      const int idx = tid + 128 * i;      // 0..511 16-byte chunks
      const int row = idx >> 3;
      const int c8  = idx & 7;
      async_ld_b128(&sK[row * ALD + c8 * 8], kp + (long)(kk + row) * HD + c8 * 8);
      async_ld_b128(&sV[row * ALD + c8 * 8], vp + (long)row * SEQ + kk + c8 * 8);
    }
    if (tid < 64) sMask[tid] = ((float)mp[kk + tid] - 1.0f) * 1e8f;
    wait_async0();
    __syncthreads();

    // scores: 16q x 64k per wave = 4 n-tiles x 2 k-steps
    v8f sf[4];
#pragma unroll
    for (int j = 0; j < 4; ++j) {
#pragma unroll
      for (int v = 0; v < 8; ++v) sf[j][v] = 0.0f;
      const v16h b0 = frag_b_ld(sK + (j * 16) * ALD,      ALD, lane);
      const v16h b1 = frag_b_ld(sK + (j * 16) * ALD + 32, ALD, lane);
      sf[j] = wmma16(aq0, b0, sf[j]);
      sf[j] = wmma16(aq1, b1, sf[j]);
    }
#pragma unroll
    for (int j = 0; j < 4; ++j)
#pragma unroll
      for (int v = 0; v < 8; ++v)
        sS[(w * 16 + hi * 8 + v) * SLD + j * 16 + nl] = sf[j][v];
    wait_ds0();

    // online softmax over this wave's 16 rows
    for (int r = 0; r < 16; ++r) {
      const int row = w * 16 + r;
      const float x1 = sS[row * SLD + lane]      + sMask[lane];
      const float x2 = sS[row * SLD + lane + 32] + sMask[lane + 32];
      float mx = fmaxf(x1, x2);
      mx = fmaxf(mx, __shfl_xor(mx, 16, 32));
      mx = fmaxf(mx, __shfl_xor(mx,  8, 32));
      mx = fmaxf(mx, __shfl_xor(mx,  4, 32));
      mx = fmaxf(mx, __shfl_xor(mx,  2, 32));
      mx = fmaxf(mx, __shfl_xor(mx,  1, 32));
      const float mold = sM[row];
      const float mnew = fmaxf(mold, mx);
      const float p1 = __expf(x1 - mnew);
      const float p2 = __expf(x2 - mnew);
      float ps = p1 + p2;
      ps += __shfl_xor(ps, 16, 32);
      ps += __shfl_xor(ps,  8, 32);
      ps += __shfl_xor(ps,  4, 32);
      ps += __shfl_xor(ps,  2, 32);
      ps += __shfl_xor(ps,  1, 32);
      const float al = __expf(mold - mnew);
      if (lane == 0) { sM[row] = mnew; sL[row] = sL[row] * al + ps; sAl[row] = al; }
      sP[row * ALD + lane]      = (_Float16)p1;
      sP[row * ALD + lane + 32] = (_Float16)p2;
    }
    wait_ds0();

    // rescale accumulators by alpha[row]
    float al8[8];
#pragma unroll
    for (int v = 0; v < 8; ++v) al8[v] = sAl[w * 16 + hi * 8 + v];
#pragma unroll
    for (int j = 0; j < 4; ++j)
#pragma unroll
      for (int v = 0; v < 8; ++v) o[j][v] *= al8[v];

    // P @ V: contraction over 64 keys (2 k-steps), 4 d-tiles
    const v16h ap0 = frag_a_ld(sP + (w * 16) * ALD,      ALD, lane);
    const v16h ap1 = frag_a_ld(sP + (w * 16) * ALD + 32, ALD, lane);
#pragma unroll
    for (int j = 0; j < 4; ++j) {
      const v16h b0 = frag_b_ld(sV + (j * 16) * ALD,      ALD, lane);
      const v16h b1 = frag_b_ld(sV + (j * 16) * ALD + 32, ALD, lane);
      o[j] = wmma16(ap0, b0, o[j]);
      o[j] = wmma16(ap1, b1, o[j]);
    }
  }

  wait_ds0();
  float li[8];
#pragma unroll
  for (int v = 0; v < 8; ++v) li[v] = 1.0f / sL[w * 16 + hi * 8 + v];

#pragma unroll
  for (int j = 0; j < 4; ++j)
#pragma unroll
    for (int v = 0; v < 8; ++v) {
      const int gq = q0 + w * 16 + hi * 8 + v;
      const int d  = j * 16 + nl;
      O[((long)(b * SEQ + gq) * NHEAD + h) * HD + d] = (_Float16)(o[j][v] * li[v]);
    }
}

// ---------------------------------------------------------------------------
// Kernel 3: output projection  Y[m][n] = sum_k A[m][k] * Wo[n][k]
// A: f16 [8192x1024] (attention output, row-major (B,S,D)) staged via async
// global->LDS copies; Wo f32 converted in-flight; Y f32.
// ---------------------------------------------------------------------------
__global__ __launch_bounds__(256)
void oproj_gemm_kernel(const _Float16* __restrict__ A, const float* __restrict__ W,
                       float* __restrict__ Y)
{
  __shared__ alignas(16) _Float16 sA[G_BM * G_LD];
  __shared__ alignas(16) _Float16 sB[G_BN * G_LD];

  const int tid  = threadIdx.x;
  const int lane = tid & 31;
  const int w    = tid >> 5;
  const int wm   = w & 3;
  const int wn   = w >> 2;
  const int m0   = blockIdx.x * G_BM;
  const int n0   = blockIdx.y * G_BN;

  v8f acc[2][4];
#pragma unroll
  for (int i = 0; i < 2; ++i)
#pragma unroll
    for (int j = 0; j < 4; ++j)
#pragma unroll
      for (int v = 0; v < 8; ++v) acc[i][j][v] = 0.0f;

  for (int k0 = 0; k0 < DMODEL; k0 += G_BK) {
    // A: 128x64 f16 = 1024 16B chunks via async global->LDS (4 per thread)
#pragma unroll
    for (int i = 0; i < 4; ++i) {
      const int idx = tid + 256 * i;
      const int row = idx >> 3;             // 8 chunks of 8 halves per row
      const int c8  = idx & 7;
      async_ld_b128(&sA[row * G_LD + c8 * 8],
                    A + (long)(m0 + row) * DMODEL + k0 + c8 * 8);
    }
    // W: f32 -> f16 through VGPRs
#pragma unroll
    for (int i = 0; i < 8; ++i) {
      const int idx = tid + 256 * i;
      const int row = idx >> 4;
      const int c4  = idx & 15;
      const float4 bv = *(const float4*)(W + (long)(n0 + row) * DMODEL + k0 + c4 * 4);
      v4h bh = { (_Float16)bv.x, (_Float16)bv.y, (_Float16)bv.z, (_Float16)bv.w };
      *(v4h*)(&sB[row * G_LD + c4 * 4]) = bh;
      if (k0 + G_BK < DMODEL)
        __builtin_prefetch(W + (long)(n0 + row) * DMODEL + k0 + G_BK + c4 * 4, 0, 3);
    }
    wait_async0();
    __syncthreads();

#pragma unroll
    for (int ks = 0; ks < 2; ++ks) {
      const v16h a0 = frag_a_ld(sA + (wm * 32 +  0) * G_LD + ks * 32, G_LD, lane);
      const v16h a1 = frag_a_ld(sA + (wm * 32 + 16) * G_LD + ks * 32, G_LD, lane);
#pragma unroll
      for (int j = 0; j < 4; ++j) {
        const v16h bf = frag_b_ld(sB + (wn * 64 + j * 16) * G_LD + ks * 32, G_LD, lane);
        acc[0][j] = wmma16(a0, bf, acc[0][j]);
        acc[1][j] = wmma16(a1, bf, acc[1][j]);
      }
    }
    __syncthreads();
  }

  const int hi = lane >> 4, nl = lane & 15;
#pragma unroll
  for (int i = 0; i < 2; ++i)
#pragma unroll
    for (int j = 0; j < 4; ++j)
#pragma unroll
      for (int v = 0; v < 8; ++v) {
        const int gm = m0 + wm * 32 + i * 16 + hi * 8 + v;
        const int gn = n0 + wn * 64 + j * 16 + nl;
        Y[(long)gm * DMODEL + gn] = acc[i][j][v];
      }
}

// ---------------------------------------------------------------------------
// Kernel 4: residual add + LayerNorm per row (1024 cols, 256 threads x 4)
// ---------------------------------------------------------------------------
__global__ __launch_bounds__(256)
void ln_kernel(const float* __restrict__ Y, const float* __restrict__ R,
               const float* __restrict__ gamma, const float* __restrict__ beta,
               float* __restrict__ out)
{
  __shared__ float red[16];
  const int row = blockIdx.x;
  const int tid = threadIdx.x;
  const int c   = tid * 4;

  const float4 y = *(const float4*)(Y + (long)row * DMODEL + c);
  const float4 r = *(const float4*)(R + (long)row * DMODEL + c);
  const float x0 = y.x + r.x, x1 = y.y + r.y, x2 = y.z + r.z, x3 = y.w + r.w;

  float s  = x0 + x1 + x2 + x3;
  float sq = x0 * x0 + x1 * x1 + x2 * x2 + x3 * x3;
  for (int off = 16; off >= 1; off >>= 1) {
    s  += __shfl_xor(s,  off, 32);
    sq += __shfl_xor(sq, off, 32);
  }
  const int w = tid >> 5, lane = tid & 31;
  if (lane == 0) { red[w] = s; red[8 + w] = sq; }
  __syncthreads();
  float ts = 0.0f, tq = 0.0f;
#pragma unroll
  for (int i = 0; i < 8; ++i) { ts += red[i]; tq += red[8 + i]; }

  const float mu   = ts * (1.0f / DMODEL);
  const float var  = tq * (1.0f / DMODEL) - mu * mu;
  const float rstd = rsqrtf(var + 1e-5f);

  const float4 g  = *(const float4*)(gamma + c);
  const float4 bb = *(const float4*)(beta + c);
  float4 ov;
  ov.x = (x0 - mu) * rstd * g.x + bb.x;
  ov.y = (x1 - mu) * rstd * g.y + bb.y;
  ov.z = (x2 - mu) * rstd * g.z + bb.z;
  ov.w = (x3 - mu) * rstd * g.w + bb.w;
  *(float4*)(out + (long)row * DMODEL + c) = ov;
}

// ---------------------------------------------------------------------------
extern "C" void kernel_launch(void* const* d_in, const int* in_sizes, int n_in,
                              void* d_out, int out_size, void* d_ws, size_t ws_size,
                              hipStream_t stream) {
  const float* q     = (const float*)d_in[0];
  const float* k     = (const float*)d_in[1];
  const float* v     = (const float*)d_in[2];
  const int*   mask  = (const int*)  d_in[3];
  const float* Wq    = (const float*)d_in[4];
  const float* Wk    = (const float*)d_in[5];
  const float* Wv    = (const float*)d_in[6];
  const float* Wo    = (const float*)d_in[7];
  const float* gamma = (const float*)d_in[8];
  const float* beta  = (const float*)d_in[9];
  float* out = (float*)d_out;

  char* ws = (char*)d_ws;
  const size_t hm = (size_t)MTOT * DMODEL * sizeof(_Float16);   // 16 MiB
  _Float16* Qh  = (_Float16*)(ws);
  _Float16* Kh  = (_Float16*)(ws + hm);
  _Float16* Vth = (_Float16*)(ws + 2 * hm);
  _Float16* Ah  = (_Float16*)(ws + 3 * hm);
  float*    Yf  = (float*)   (ws + 4 * hm);

  const dim3 gg(MTOT / G_BM, DMODEL / G_BN);   // 64 x 8
  const dim3 gb(256);

  proj_gemm_kernel<<<gg, gb, 0, stream>>>(q, Wq, Qh, 0, 0.125f);  // 1/sqrt(64)
  proj_gemm_kernel<<<gg, gb, 0, stream>>>(k, Wk, Kh, 0, 1.0f);
  proj_gemm_kernel<<<gg, gb, 0, stream>>>(v, Wv, Vth, 2, 1.0f);

  attn_kernel<<<dim3(SEQ / 64, BATCH * NHEAD), dim3(128), 0, stream>>>(Qh, Kh, Vth, mask, Ah);

  oproj_gemm_kernel<<<gg, gb, 0, stream>>>(Ah, Wo, Yf);

  ln_kernel<<<dim3(MTOT), dim3(256), 0, stream>>>(Yf, q, gamma, beta, out);
}
